// EdgeConvBlock_89945205112954
// MI455X (gfx1250) — compile-verified, matrix operand order checked by reading, not run on hardware
//
#include <hip/hip_runtime.h>
#include <hip/hip_bf16.h>

typedef __attribute__((ext_vector_type(16))) __bf16 v16bf;
typedef __attribute__((ext_vector_type(8)))  float  v8f;

#define BB   16
#define CIN  128
#define NPTS 2048
#define COUT 256
#define KNN  20

// Load one 16x32 bf16 WMMA fragment half-pair for this lane from a contiguous
// "point-major" row: elements [k0..k0+7] and [k0+16..k0+23] as two b128 loads.
union fragu { v16bf v; uint4 q[2]; };
__device__ __forceinline__ v16bf load_frag(const __bf16* row, int k0) {
  fragu f;
  f.q[0] = *(const uint4*)(row + k0);
  f.q[1] = *(const uint4*)(row + k0 + 16);
  return f.v;
}

// ---------------------------------------------------------------------------
// Kernel 1: x (B,C,N) fp32 -> point-major bf16 xp[b][n][c] + squared norms.
// Packed 16B stores; per-thread one point.
// ---------------------------------------------------------------------------
__global__ __launch_bounds__(256) void prep_x_kernel(const float* __restrict__ x,
                                                     __bf16* __restrict__ xp,
                                                     float* __restrict__ sq) {
  int g = blockIdx.x * 256 + threadIdx.x;        // 0 .. B*N-1
  int b = g >> 11, n = g & (NPTS - 1);
  const float* col = x + (size_t)b * CIN * NPTS + n;
  __bf16* out = xp + (size_t)g * CIN;
  float s = 0.f;
  #pragma unroll
  for (int c0 = 0; c0 < CIN; c0 += 8) {
    fragu u;
    __bf16* h = (__bf16*)&u;
    #pragma unroll
    for (int j = 0; j < 8; ++j) {
      float v = col[(size_t)(c0 + j) * NPTS];
      s += v * v;
      h[j] = (__bf16)v;
    }
    *(uint4*)(out + c0) = u.q[0];
  }
  sq[g] = s;
}

// ---------------------------------------------------------------------------
// Kernel 2: W (256 x 256) -> Wd = W1 - W2 and W2, bf16 row-major [256][128]
// ---------------------------------------------------------------------------
__global__ __launch_bounds__(256) void prep_w_kernel(const float* __restrict__ W,
                                                     __bf16* __restrict__ wd,
                                                     __bf16* __restrict__ w2) {
  int i = blockIdx.x * 256 + threadIdx.x;        // 0 .. 256*128-1
  int o = i >> 7, c = i & 127;
  float w1 = W[o * 2 * CIN + c];
  float wB = W[o * 2 * CIN + CIN + c];
  wd[i] = (__bf16)(w1 - wB);
  w2[i] = (__bf16)wB;
}

// ---------------------------------------------------------------------------
// Kernel 3: kNN. One block = (batch b, 16 query rows). WMMA bf16 inner
// products vs all 2048 points. A-fragments (queries) are register-cached for
// the whole block; B-fragments are wide b128 loads from point-major xp.
// 16x2048 fp32 distance block lives in LDS (128KB of 320KB), then
// wave-parallel iterative top-20 with wave32 butterflies.
// ---------------------------------------------------------------------------
__global__ __launch_bounds__(256) void knn_kernel(const __bf16* __restrict__ xp,
                                                  const float* __restrict__ sq,
                                                  int* __restrict__ knn) {
  __shared__ float sqq[16];
  __shared__ float pd[16 * NPTS];    // 128KB distance row-block
  const int b   = blockIdx.y;
  const int q0  = blockIdx.x * 16;
  const int tid = threadIdx.x;
  const int lane = tid & 31, wid = tid >> 5;
  const int lc = lane & 15, rh = lane >> 4;

  // register-cache A fragments: query row = lane%16, K split by lane half
  const __bf16* qrow = xp + (size_t)(b * NPTS + q0 + lc) * CIN;
  v16bf afr[4];
  #pragma unroll
  for (int ks = 0; ks < 4; ++ks) afr[ks] = load_frag(qrow, ks * 32 + rh * 8);

  if (tid < 16) sqq[tid] = sq[b * NPTS + q0 + tid];
  __syncthreads();

  // each of the 8 waves covers 16 column tiles (16 cols each)
  for (int i = 0; i < 16; ++i) {
    int c0 = (wid * 16 + i) * 16;
    const __bf16* crow = xp + (size_t)(b * NPTS + c0 + lc) * CIN;  // col = lane%16
    v8f acc = {};
    #pragma unroll
    for (int ks = 0; ks < 4; ++ks) {                 // K = 128 in 4 x 32 steps
      v16bf bm = load_frag(crow, ks * 32 + rh * 8);
      acc = __builtin_amdgcn_wmma_f32_16x16x32_bf16(false, afr[ks], false, bm,
                                                    (short)0, acc, false, false);
    }
    float sqc = sq[b * NPTS + c0 + lc];
    #pragma unroll
    for (int r = 0; r < 8; ++r) {
      int qr = r + 8 * rh;                           // C/D: M = r + 8*(lane/16)
      pd[qr * NPTS + c0 + lc] = sqq[qr] - 2.0f * acc[r] + sqc;
    }
  }
  __syncthreads();

  // iterative top-20: each wave owns rows wid and wid+8
  for (int row = wid; row < 16; row += 8) {
    float* prow = pd + row * NPTS;
    int gq = b * NPTS + q0 + row;
    for (int t = 0; t < KNN; ++t) {
      float best = 3.0e38f; int bidx = 0;
      for (int c = lane; c < NPTS; c += 32) {
        float v = prow[c];
        if (v < best || (v == best && c < bidx)) { best = v; bidx = c; }
      }
      #pragma unroll
      for (int off = 16; off > 0; off >>= 1) {       // wave32 butterfly min
        float ov = __shfl_xor(best, off, 32);
        int   oi = __shfl_xor(bidx, off, 32);
        if (ov < best || (ov == best && oi < bidx)) { best = ov; bidx = oi; }
      }
      if (lane == (bidx & 31)) prow[bidx] = 3.0e38f; // knock out winner
      if (lane == 0) knn[(size_t)gq * KNN + t] = bidx;
    }
  }
}

// ---------------------------------------------------------------------------
// Kernel 4: out[b,o,n] = sum_c Wm[o,c] * x[b,c,n]  (M=256, K=128, N=2048/b).
// No LDS: A (W rows) register-cached per wave (2 M-tiles x 4 K-steps),
// B straight from point-major xp as b128 loads (L2-resident, 8MB).
// ---------------------------------------------------------------------------
__global__ __launch_bounds__(256) void gemm_kernel(const __bf16* __restrict__ Wm,
                                                   const __bf16* __restrict__ xp,
                                                   float* __restrict__ out) {
  const int b  = blockIdx.y;
  const int n0 = blockIdx.x * 128;
  const int tid = threadIdx.x;
  const int lane = tid & 31, wid = tid >> 5;
  const int lc = lane & 15, rh = lane >> 4;

  // preload A fragments: this wave's two M-tiles, all 4 K-steps (64 VGPRs)
  v16bf afr[2][4];
  #pragma unroll
  for (int mt = 0; mt < 2; ++mt) {
    const __bf16* wrow = Wm + (size_t)((wid * 2 + mt) * 16 + lc) * CIN;
    #pragma unroll
    for (int ks = 0; ks < 4; ++ks) afr[mt][ks] = load_frag(wrow, ks * 32 + rh * 8);
  }

  for (int nt = 0; nt < 8; ++nt) {
    const __bf16* xrow = xp + (size_t)(b * NPTS + n0 + nt * 16 + lc) * CIN;
    if (nt + 1 < 8)
      __builtin_prefetch(xrow + 16 * CIN, 0, 1);     // global_prefetch next tile
    v8f acc0 = {}, acc1 = {};
    #pragma unroll
    for (int ks = 0; ks < 4; ++ks) {
      v16bf bm = load_frag(xrow, ks * 32 + rh * 8);
      acc0 = __builtin_amdgcn_wmma_f32_16x16x32_bf16(false, afr[0][ks], false, bm,
                                                     (short)0, acc0, false, false);
      acc1 = __builtin_amdgcn_wmma_f32_16x16x32_bf16(false, afr[1][ks], false, bm,
                                                     (short)0, acc1, false, false);
    }
    int n = n0 + nt * 16 + lc;
    #pragma unroll
    for (int r = 0; r < 8; ++r) {
      int o0 = (wid * 2) * 16 + r + 8 * rh;
      int o1 = (wid * 2 + 1) * 16 + r + 8 * rh;
      out[((size_t)b * COUT + o0) * NPTS + n] = acc0[r];
      out[((size_t)b * COUT + o1) * NPTS + n] = acc1[r];
    }
  }
}

// ---------------------------------------------------------------------------
// Kernel 5: BN partial stats. y = u[b,o,n] + v[b,o,knn[b,n,j]] (never stored).
// Block = 256 threads (one per channel o), 32 points per block.
// Deterministic: per-block partials, summed in order in kernel 6.
// ---------------------------------------------------------------------------
__global__ __launch_bounds__(256) void stats_kernel(const float* __restrict__ u,
                                                    const float* __restrict__ v,
                                                    const int* __restrict__ knn,
                                                    float* __restrict__ part) {
  __shared__ int kn[KNN];
  const int o = threadIdx.x;
  float s0 = 0.f, s1 = 0.f;
  for (int p = 0; p < 32; ++p) {
    int g = blockIdx.x * 32 + p;
    int b = g >> 11, n = g & (NPTS - 1);
    __syncthreads();
    if (o < KNN) kn[o] = knn[(size_t)g * KNN + o];
    __syncthreads();
    float uv = u[((size_t)b * COUT + o) * NPTS + n];
    const float* vb = v + ((size_t)b * COUT + o) * NPTS;
    #pragma unroll 4
    for (int j = 0; j < KNN; ++j) {
      float y = uv + vb[kn[j]];
      s0 += y; s1 += y * y;
    }
  }
  part[(size_t)blockIdx.x * 512 + o]       = s0;
  part[(size_t)blockIdx.x * 512 + 256 + o] = s1;
}

__global__ __launch_bounds__(512) void reduce_kernel(const float* __restrict__ part,
                                                     float* __restrict__ sums) {
  int t = threadIdx.x;                               // 0..511 (sum | sumsq)
  float a = 0.f;
  for (int i = 0; i < 1024; ++i) a += part[(size_t)i * 512 + t];
  sums[t] = a;
}

__global__ __launch_bounds__(256) void scale_kernel(const float* __restrict__ sums,
                                                    const float* __restrict__ gamma,
                                                    const float* __restrict__ beta,
                                                    float* __restrict__ sc,
                                                    float* __restrict__ sh) {
  int o = threadIdx.x;
  const float inv = 1.0f / ((float)BB * NPTS * KNN);
  float m   = sums[o] * inv;
  float var = sums[256 + o] * inv - m * m;
  float s   = gamma[o] * rsqrtf(var + 1e-5f);
  sc[o] = s;
  sh[o] = beta[o] - s * m;
}

// ---------------------------------------------------------------------------
// Kernel 7: out[b,o,n] = relu(max_j (s*(u+v_j)+t)) = relu(s*(u+ext_j v)+t)
// with ext = max if s>=0 else min (monotone affine commutes with max).
// ---------------------------------------------------------------------------
__global__ __launch_bounds__(256) void out_kernel(const float* __restrict__ u,
                                                  const float* __restrict__ v,
                                                  const int* __restrict__ knn,
                                                  const float* __restrict__ sc,
                                                  const float* __restrict__ sh,
                                                  float* __restrict__ out) {
  __shared__ int kn[KNN];
  const int o = threadIdx.x;
  for (int p = 0; p < 8; ++p) {
    int g = blockIdx.x * 8 + p;
    int b = g >> 11, n = g & (NPTS - 1);
    __syncthreads();
    if (o < KNN) kn[o] = knn[(size_t)g * KNN + o];
    __syncthreads();
    const float* vb = v + ((size_t)b * COUT + o) * NPTS;
    float mx = -3.0e38f, mn = 3.0e38f;
    #pragma unroll 4
    for (int j = 0; j < KNN; ++j) {
      float vv = vb[kn[j]];
      mx = fmaxf(mx, vv); mn = fminf(mn, vv);
    }
    float uv = u[((size_t)b * COUT + o) * NPTS + n];
    float s = sc[o], t = sh[o];
    float yext = uv + (s >= 0.0f ? mx : mn);
    out[((size_t)b * COUT + o) * NPTS + n] = fmaxf(0.0f, s * yext + t);
  }
}

// ---------------------------------------------------------------------------
extern "C" void kernel_launch(void* const* d_in, const int* in_sizes, int n_in,
                              void* d_out, int out_size, void* d_ws, size_t ws_size,
                              hipStream_t stream) {
  const float* x     = (const float*)d_in[0];
  const float* W     = (const float*)d_in[1];
  const float* gamma = (const float*)d_in[2];
  const float* beta  = (const float*)d_in[3];
  // d_in[4] is k == 20 (compile-time constant here)

  // workspace layout (all offsets 256B aligned), ~80.5 MB total
  constexpr size_t SZ_XP   = (size_t)BB * NPTS * CIN * 2;   //  8,388,608
  constexpr size_t SZ_SQ   = (size_t)BB * NPTS * 4;         //    131,072
  constexpr size_t SZ_W    = (size_t)COUT * CIN * 2;        //     65,536
  constexpr size_t SZ_KNN  = (size_t)BB * NPTS * KNN * 4;   //  2,621,440
  constexpr size_t SZ_UV   = (size_t)BB * COUT * NPTS * 4;  // 33,554,432
  constexpr size_t SZ_PART = (size_t)1024 * 512 * 4;        //  2,097,152

  char* ws = (char*)d_ws;
  size_t off = 0;
  __bf16* xp  = (__bf16*)(ws + off); off += SZ_XP;
  float*  sq  = (float*)(ws + off);  off += SZ_SQ;
  __bf16* wd  = (__bf16*)(ws + off); off += SZ_W;
  __bf16* w2  = (__bf16*)(ws + off); off += SZ_W;
  int*    knn = (int*)(ws + off);    off += SZ_KNN;
  float*  u   = (float*)(ws + off);  off += SZ_UV;
  float*  v   = (float*)(ws + off);  off += SZ_UV;
  float*  part= (float*)(ws + off);  off += SZ_PART;
  float*  sums= (float*)(ws + off);  off += 512 * 4;   // sums[256] | sumsq[256]
  float*  sc  = (float*)(ws + off);  off += 256 * 4;
  float*  sh  = (float*)(ws + off);  off += 256 * 4;

  prep_x_kernel<<<BB * NPTS / 256, 256, 0, stream>>>(x, xp, sq);
  prep_w_kernel<<<COUT * CIN / 256, 256, 0, stream>>>(W, wd, w2);
  knn_kernel<<<dim3(NPTS / 16, BB), 256, 0, stream>>>(xp, sq, knn);
  gemm_kernel<<<dim3(NPTS / 128, BB), 256, 0, stream>>>(wd, xp, u);
  gemm_kernel<<<dim3(NPTS / 128, BB), 256, 0, stream>>>(w2, xp, v);
  stats_kernel<<<BB * NPTS / 32, 256, 0, stream>>>(u, v, knn, part);
  reduce_kernel<<<1, 512, 0, stream>>>(part, sums);
  scale_kernel<<<1, 256, 0, stream>>>(sums, gamma, beta, sc, sh);
  out_kernel<<<BB * NPTS / 8, 256, 0, stream>>>(u, v, knn, sc, sh, (float*)d_out);
}
